// MaSA_9363028705991
// MI455X (gfx1250) — compile-verified, matrix operand order checked by reading, not run on hardware
//
#include <hip/hip_runtime.h>
#include <math.h>

// ---------------------------------------------------------------------------
// MaSA (2D decomposed retention) for MI455X / gfx1250, wave32 + WMMA f16.
// All matmuls (QKV GEMMs, QK^T, P@V both axes, output GEMM) run on
// v_wmma_f32_16x16x32_f16 with f32 accumulation. Every WMMA operand tile is
// staged in LDS K-minor so fragments load as two ds_load_b128 per operand.
// ---------------------------------------------------------------------------

typedef __attribute__((ext_vector_type(16))) _Float16 v16h;
typedef __attribute__((ext_vector_type(8)))  _Float16 v8h;
typedef __attribute__((ext_vector_type(8)))  float    v8f;

constexpr int Bsz  = 16;
constexpr int Cdim = 256;
constexpr int Hdim = 64;
constexpr int Wdim = 64;
constexpr int NH   = 4;
constexpr int HW   = Hdim * Wdim;          // 4096
constexpr int Mrows = Bsz * HW;            // 65536

// ---------------- WMMA fragment helper (CDNA5 16-bit layout) ---------------
// Per-lane data: row r0+(lane&15), K runs {kb..kb+7} and {kb+16..kb+23} with
// kb = k0 + (lane>=16 ? 8 : 0). Tiles are stored K-minor, so each run is one
// 16-byte LDS load. Serves A (row-major) and B (stored n-major / transposed).
__device__ __forceinline__ v16h frag_ld(const _Float16* p, int ld, int r0, int k0) {
    const int lane = threadIdx.x & 31;
    const _Float16* base =
        p + (size_t)(r0 + (lane & 15)) * ld + k0 + ((lane >> 4) << 3);
    const v8h lo = *(const v8h*)(base);        // K = kb .. kb+7
    const v8h hi = *(const v8h*)(base + 16);   // K = kb+16 .. kb+23
    v16h f;
#pragma unroll
    for (int e = 0; e < 8; ++e) { f[e] = lo[e]; f[e + 8] = hi[e]; }
    return f;
}

__device__ __forceinline__ v8f wmma_f16(v16h a, v16h b, v8f c) {
    return __builtin_amdgcn_wmma_f32_16x16x32_f16(false, a, false, b,
                                                  (short)0, c, false, false);
}

// ---------------- TDM probe (architected NULL tensor: moves no data) -------
#if __has_builtin(__builtin_amdgcn_tensor_load_to_lds)
typedef unsigned int u32x4 __attribute__((ext_vector_type(4)));
typedef int          i32x8 __attribute__((ext_vector_type(8)));
typedef int          i32x4 __attribute__((ext_vector_type(4)));
__device__ __forceinline__ void tdm_null_probe() {
    u32x4 g0 = {};   // D# group0: count[1:0]=0 -> NULL tensor descriptor
    i32x8 g1 = {};
    i32x4 g2 = {};
    i32x4 g3 = {};
    i32x8 g4 = {};
    // 6-arg form (clang-23 / therock-10.0 headers)
    __builtin_amdgcn_tensor_load_to_lds(g0, g1, g2, g3, g4, 0);
    __builtin_amdgcn_s_wait_tensorcnt(0);
}
#endif

// ---------------- pack kernels ---------------------------------------------
// x NCHW f32 -> A16 (row = b*HW + h*W + w, col = c) f16
__global__ void pack_x_kernel(const float* __restrict__ x, _Float16* __restrict__ A16) {
    const int idx = blockIdx.x * blockDim.x + threadIdx.x;
    if (idx >= Mrows * Cdim) return;
    const int c   = idx & (Cdim - 1);
    const int row = idx >> 8;
    const int b   = row >> 12;
    const int hw  = row & (HW - 1);
    A16[idx] = (_Float16)x[(size_t)(b * Cdim + c) * HW + hw];
}

// Wq|Wk|Wv|Wo (K x N row-major f32) -> f16 TRANSPOSED: W16T[wsel][n][k]
__global__ void pack_w_kernel(const float* __restrict__ Wq, const float* __restrict__ Wk,
                              const float* __restrict__ Wv, const float* __restrict__ Wo,
                              _Float16* __restrict__ W16T) {
#if __has_builtin(__builtin_amdgcn_tensor_load_to_lds)
    if (blockIdx.x == 0 && threadIdx.x < 32) tdm_null_probe();
#endif
    const int idx = blockIdx.x * blockDim.x + threadIdx.x;
    if (idx >= 4 * Cdim * Cdim) return;
    const int wsel = idx >> 16;
    const int rem  = idx & 65535;
    const int n = rem >> 8;
    const int k = rem & (Cdim - 1);
    const float* src = (wsel == 0) ? Wq : (wsel == 1) ? Wk : (wsel == 2) ? Wv : Wo;
    W16T[idx] = (_Float16)src[k * Cdim + n];
}

// ---------------- WMMA GEMM: C(f32, Mx256) = A(f16) * B(f16) ---------------
// B supplied transposed (Bt[n][k]). block = 256 threads (8 waves),
// tile 128x64, K-step 32. LDS ld = 40 halves = 80B: 16B aligned rows,
// conflict-free across 16 lanes.
__global__ __launch_bounds__(256)
void gemm_f16_kernel(const _Float16* __restrict__ A, const _Float16* __restrict__ Bt,
                     float* __restrict__ C) {
    __shared__ __align__(16) _Float16 As[128][40];
    __shared__ __align__(16) _Float16 BsT[64][40];   // [n][k]
    const int t  = threadIdx.x;
    const int m0 = blockIdx.x * 128;
    const int n0 = blockIdx.y * 64;
    const int wid = t >> 5;
    const int wm = wid & 3;      // 0..3 -> 32-row slab
    const int wn = wid >> 2;     // 0..1 -> 32-col slab
    v8f acc[2][2] = {};

    for (int k0 = 0; k0 < Cdim; k0 += 32) {
        {   // stage A tile 128x32: 2 x b128 per thread
            const int r = t >> 1, cb = (t & 1) * 16;
            const v8h* src = (const v8h*)&A[(size_t)(m0 + r) * Cdim + k0 + cb];
            *(v8h*)&As[r][cb]     = src[0];
            *(v8h*)&As[r][cb + 8] = src[1];
            if (k0 + 32 < Cdim) __builtin_prefetch(src + 4, 0, 1); // next K tile
        }
        {   // stage B tile 64x32 (n-major): 1 x b128 per thread
            const int n = t >> 2, cb = (t & 3) * 8;
            *(v8h*)&BsT[n][cb] =
                *(const v8h*)&Bt[(size_t)(n0 + n) * Cdim + k0 + cb];
        }
        __syncthreads();
        v16h a0 = frag_ld(&As[0][0],  40, wm * 32,      0);
        v16h a1 = frag_ld(&As[0][0],  40, wm * 32 + 16, 0);
        v16h b0 = frag_ld(&BsT[0][0], 40, wn * 32,      0);
        v16h b1 = frag_ld(&BsT[0][0], 40, wn * 32 + 16, 0);
        acc[0][0] = wmma_f16(a0, b0, acc[0][0]);
        acc[0][1] = wmma_f16(a0, b1, acc[0][1]);
        acc[1][0] = wmma_f16(a1, b0, acc[1][0]);
        acc[1][1] = wmma_f16(a1, b1, acc[1][1]);
        __syncthreads();
    }

    const int lane = t & 31;
    const int hi = lane >> 4, nn = lane & 15;
#pragma unroll
    for (int i = 0; i < 2; ++i)
#pragma unroll
        for (int j = 0; j < 2; ++j)
#pragma unroll
            for (int r = 0; r < 8; ++r) {
                const int row = m0 + wm * 32 + i * 16 + hi * 8 + r;
                const int col = n0 + wn * 32 + j * 16 + nn;
                C[(size_t)row * Cdim + col] = acc[i][j][r];
            }
}

// ---------------- fused bias + k-scale + RoPE(theta_shift) + f16 pack ------
// qr16/kr16: [row][c] f16.  v: bias-added f32 in place (LEPE) and f16
// TRANSPOSED copy v16T[b][h][n][d][w] for vectorized attention staging.
__global__ void rope_pack_kernel(const float* __restrict__ qraw, const float* __restrict__ kraw,
                                 float* __restrict__ vraw,
                                 const float* __restrict__ bq, const float* __restrict__ bk,
                                 const float* __restrict__ bv,
                                 _Float16* __restrict__ qr16, _Float16* __restrict__ kr16,
                                 _Float16* __restrict__ v16T) {
    const int idx = blockIdx.x * blockDim.x + threadIdx.x;
    if (idx >= Mrows * (Cdim / 2)) return;
    const int pair = idx & 127;
    const int row  = idx >> 7;
    const int c0   = pair * 2;
    const int dd   = c0 & 63;          // channel within head (d = 64)
    const int i    = dd >> 1;          // 0..31 frequency index
    const int pos  = row & (HW - 1);   // flattened (h,w) position
    const float angle = powf(10000.f, -(float)i * (1.0f / 31.0f));
    const float ang = (float)pos * angle;
    const float s = sinf(ang), cz = cosf(ang);
    const size_t o = (size_t)row * Cdim + c0;

    const float q0 = qraw[o] + bq[c0], q1 = qraw[o + 1] + bq[c0 + 1];
    qr16[o]     = (_Float16)(q0 * cz - q1 * s);
    qr16[o + 1] = (_Float16)(q1 * cz + q0 * s);

    const float k0 = (kraw[o] + bk[c0]) * 0.125f;       // d^-0.5
    const float k1 = (kraw[o + 1] + bk[c0 + 1]) * 0.125f;
    kr16[o]     = (_Float16)(k0 * cz - k1 * s);
    kr16[o + 1] = (_Float16)(k1 * cz + k0 * s);

    const float v0 = vraw[o] + bv[c0], v1 = vraw[o + 1] + bv[c0 + 1];
    vraw[o] = v0; vraw[o + 1] = v1;                     // f32 (for LEPE conv)
    // transposed f16 copy: [b][h][n][d][w]
    const int b = row >> 12, hw = row & (HW - 1);
    const int h = hw >> 6, w = hw & 63;
    const int n = c0 >> 6;
    const size_t tb = ((((size_t)(b * 64 + h)) * NH + n) * 64 + dd) * 64 + w;
    v16T[tb]      = (_Float16)v0;
    v16T[tb + 64] = (_Float16)v1;   // dd+1 -> +64 in this layout
}

// ---------------- 5x5 depthwise conv (LEPE), NHWC f32, pad 2 ---------------
__global__ void lepe_kernel(const float* __restrict__ v32, const float* __restrict__ dwk,
                            const float* __restrict__ dwb, float* __restrict__ lep) {
    const int idx = blockIdx.x * blockDim.x + threadIdx.x;
    if (idx >= Mrows * Cdim) return;
    const int c   = idx & (Cdim - 1);
    const int row = idx >> 8;
    const int b   = row >> 12;
    const int hw  = row & (HW - 1);
    const int h   = hw >> 6;
    const int w   = hw & 63;
    float acc = dwb[c];
#pragma unroll
    for (int dy = 0; dy < 5; ++dy) {
        const int hy = h + dy - 2;
        if (hy < 0 || hy >= Hdim) continue;
#pragma unroll
        for (int dx = 0; dx < 5; ++dx) {
            const int wx = w + dx - 2;
            if (wx < 0 || wx >= Wdim) continue;
            acc += v32[(size_t)(b * HW + hy * Wdim + wx) * Cdim + c] *
                   dwk[(dy * 5 + dx) * Cdim + c];
        }
    }
    lep[(size_t)row * Cdim + c] = acc;
}

// ---------------- axis attention (retention) -------------------------------
// STAGE 0: along W (block=(h,head,b)); V from v16T [b][h][n][d][w];
//          output f16 -> v5T [b][w][n][d][h].
// STAGE 1: along H (block=(w,head,b)); V from v5T  [b][w][n][d][h];
//          output f32 -> attn [b,h,w][c].
// Both V sources share layout [b][slice][n][d][pos] -> identical staging.
template <int STAGE>
__global__ __launch_bounds__(128)
void attn_kernel(const _Float16* __restrict__ qr16, const _Float16* __restrict__ kr16,
                 const _Float16* __restrict__ vsrc, void* __restrict__ outp) {
    __shared__ __align__(16) _Float16 Qs[64][72];   // [q][dchan]
    __shared__ __align__(16) _Float16 Ks[64][72];   // [key][dchan] (= n-major for QK^T)
    __shared__ __align__(16) _Float16 VsT[64][72];  // [d][key]    (= n-major for P@V)
    __shared__ __align__(16) float    Ss[64][72];
    __shared__ __align__(16) _Float16 Ps[64][72];

    const int slice = blockIdx.x;   // h (STAGE 0) or w (STAGE 1)
    const int n     = blockIdx.y;   // head
    const int b     = blockIdx.z;
    const int t     = threadIdx.x;

    {   // stage Q,K (rows of qr16/kr16) and V (rows of [d][pos] source)
        const int p  = t >> 1;           // row index 0..63
        const int db = (t & 1) * 32;     // 32-half chunk
        const size_t grow = (STAGE == 0)
            ? ((size_t)b * HW + (size_t)slice * 64 + p)
            : ((size_t)b * HW + (size_t)p * 64 + slice);
        const size_t base = grow * Cdim + n * 64 + db;
        const v8h* qsrc = (const v8h*)&qr16[base];
        const v8h* ksrc = (const v8h*)&kr16[base];
        const size_t vb = ((((size_t)(b * 64 + slice)) * NH + n) * 64 + p) * 64 + db;
        const v8h* vv = (const v8h*)&vsrc[vb];
#pragma unroll
        for (int j = 0; j < 4; ++j) {
            *(v8h*)&Qs[p][db + 8 * j]  = qsrc[j];
            *(v8h*)&Ks[p][db + 8 * j]  = ksrc[j];
            *(v8h*)&VsT[p][db + 8 * j] = vv[j];
        }
    }
    __syncthreads();

    const int wid   = t >> 5;
    const int lane  = t & 31;
    const int qbase = wid * 16;         // each wave owns 16 query rows
    const int hi = lane >> 4, nn = lane & 15;

    // S = Q @ K^T  (B[k][n] = Ks[n][k]: n-major, contiguous fragments)
    v8f sacc[4] = {};
#pragma unroll
    for (int kk = 0; kk < 64; kk += 32) {
        v16h a = frag_ld(&Qs[0][0], 72, qbase, kk);
#pragma unroll
        for (int nt = 0; nt < 4; ++nt) {
            v16h bf = frag_ld(&Ks[0][0], 72, nt * 16, kk);
            sacc[nt] = wmma_f16(a, bf, sacc[nt]);
        }
    }

    // + Manhattan decay mask: |q - k| * log(1 - 2^-(1 + 3n/4))
    const float decay = logf(1.0f - exp2f(-(1.0f + 0.75f * (float)n)));
#pragma unroll
    for (int nt = 0; nt < 4; ++nt)
#pragma unroll
        for (int r = 0; r < 8; ++r) {
            const int q = qbase + hi * 8 + r;
            const int kcol = nt * 16 + nn;
            Ss[q][kcol] = sacc[nt][r] + decay * fabsf((float)(q - kcol));
        }
    __syncthreads();

    // row softmax -> f16 P (two-pass)
    if (t < 64) {
        float mx = -1e30f;
        for (int k = 0; k < 64; ++k) mx = fmaxf(mx, Ss[t][k]);
        float sum = 0.f;
        for (int k = 0; k < 64; ++k) sum += expf(Ss[t][k] - mx);
        const float inv = 1.0f / sum;
        for (int k = 0; k < 64; ++k) Ps[t][k] = (_Float16)(expf(Ss[t][k] - mx) * inv);
    }
    __syncthreads();

    // O = P @ V  (B[k][n] = VsT[n=d][k=key]: n-major, contiguous fragments)
    v8f oacc[4] = {};
#pragma unroll
    for (int kk = 0; kk < 64; kk += 32) {
        v16h a = frag_ld(&Ps[0][0], 72, qbase, kk);
#pragma unroll
        for (int nt = 0; nt < 4; ++nt) {
            v16h bf = frag_ld(&VsT[0][0], 72, nt * 16, kk);
            oacc[nt] = wmma_f16(a, bf, oacc[nt]);
        }
    }

#pragma unroll
    for (int nt = 0; nt < 4; ++nt)
#pragma unroll
        for (int r = 0; r < 8; ++r) {
            const int q = qbase + hi * 8 + r;
            const int d = nt * 16 + nn;
            if constexpr (STAGE == 0) {
                // v5T layout [b][w=q][n][d][h=slice]
                ((_Float16*)outp)[((((size_t)(b * 64 + q)) * NH + n) * 64 + d) * 64 + slice] =
                    (_Float16)oacc[nt][r];
            } else {
                ((float*)outp)[((size_t)b * HW + (size_t)q * 64 + slice) * Cdim + n * 64 + d] =
                    oacc[nt][r];
            }
        }
}

// ---------------- (attn + lepe) -> f16 for final GEMM ----------------------
__global__ void add_pack_kernel(const float* __restrict__ attn, const float* __restrict__ lep,
                                _Float16* __restrict__ aout) {
    const int idx = blockIdx.x * blockDim.x + threadIdx.x;
    if (idx >= Mrows * Cdim) return;
    aout[idx] = (_Float16)(attn[idx] + lep[idx]);
}

// ---------------- final bias + NHWC->NCHW scatter --------------------------
__global__ void scatter_out_kernel(const float* __restrict__ Cf, const float* __restrict__ bo,
                                   float* __restrict__ out) {
    const int idx = blockIdx.x * blockDim.x + threadIdx.x;
    if (idx >= Mrows * Cdim) return;
    const int c   = idx & (Cdim - 1);
    const int row = idx >> 8;
    const int b   = row >> 12;
    const int hw  = row & (HW - 1);
    out[(size_t)(b * Cdim + c) * HW + hw] = Cf[(size_t)row * Cdim + c] + bo[c];
}

// ---------------------------------------------------------------------------
extern "C" void kernel_launch(void* const* d_in, const int* in_sizes, int n_in,
                              void* d_out, int out_size, void* d_ws, size_t ws_size,
                              hipStream_t stream) {
    const float* x   = (const float*)d_in[0];
    const float* Wq  = (const float*)d_in[1];
    const float* bq  = (const float*)d_in[2];
    const float* Wk  = (const float*)d_in[3];
    const float* bk  = (const float*)d_in[4];
    const float* Wv  = (const float*)d_in[5];
    const float* bv  = (const float*)d_in[6];
    const float* dwk = (const float*)d_in[7];
    const float* dwb = (const float*)d_in[8];
    const float* Wo  = (const float*)d_in[9];
    const float* bo  = (const float*)d_in[10];
    float* out = (float*)d_out;

    char* ws = (char*)d_ws;
    const size_t MB = 1ull << 20;
    // workspace map (stream-ordered reuse; high-water ~353 MB)
    _Float16* A16   = (_Float16*)(ws + 0);          // 32 MB  (later: aout16)
    _Float16* W16T  = (_Float16*)(ws + 32  * MB);   // 0.5 MB (transposed weights)
    float*    q_raw = (float*)   (ws + 33  * MB);   // 64 MB  (later: attn, Cfinal)
    float*    k_raw = (float*)   (ws + 97  * MB);   // 64 MB  (later: lepe)
    float*    v_raw = (float*)   (ws + 161 * MB);   // 64 MB  (bias added in place)
    _Float16* qr16  = (_Float16*)(ws + 225 * MB);   // 32 MB
    _Float16* kr16  = (_Float16*)(ws + 257 * MB);   // 32 MB
    _Float16* v16T  = (_Float16*)(ws + 289 * MB);   // 32 MB  [b][h][n][d][w]
    _Float16* v5T   = (_Float16*)(ws + 321 * MB);   // 32 MB  [b][w][n][d][h]
    float*    lepeb  = k_raw;
    float*    attnb  = q_raw;
    _Float16* aout16 = A16;
    float*    cfin   = q_raw;

    const int nElem = Mrows * Cdim;                 // 16.7M
    pack_x_kernel<<<nElem / 256, 256, 0, stream>>>(x, A16);
    pack_w_kernel<<<(4 * Cdim * Cdim) / 256, 256, 0, stream>>>(Wq, Wk, Wv, Wo, W16T);

    dim3 ggrid(Mrows / 128, Cdim / 64);             // (512, 4)
    gemm_f16_kernel<<<ggrid, 256, 0, stream>>>(A16, W16T + 0 * Cdim * Cdim, q_raw);
    gemm_f16_kernel<<<ggrid, 256, 0, stream>>>(A16, W16T + 1 * Cdim * Cdim, k_raw);
    gemm_f16_kernel<<<ggrid, 256, 0, stream>>>(A16, W16T + 2 * Cdim * Cdim, v_raw);

    rope_pack_kernel<<<(Mrows * (Cdim / 2)) / 256, 256, 0, stream>>>(
        q_raw, k_raw, v_raw, bq, bk, bv, qr16, kr16, v16T);

    lepe_kernel<<<nElem / 256, 256, 0, stream>>>(v_raw, dwk, dwb, lepeb);

    dim3 agrid(Hdim, NH, Bsz);                      // (64, 4, 16)
    attn_kernel<0><<<agrid, 128, 0, stream>>>(qr16, kr16, v16T, (void*)v5T);
    attn_kernel<1><<<agrid, 128, 0, stream>>>(qr16, kr16, v5T,  (void*)attnb);

    add_pack_kernel<<<nElem / 256, 256, 0, stream>>>(attnb, lepeb, aout16);
    gemm_f16_kernel<<<ggrid, 256, 0, stream>>>(aout16, W16T + 3 * Cdim * Cdim, cfin);
    scatter_out_kernel<<<nElem / 256, 256, 0, stream>>>(cfin, bo, out);
}
